// GridGNNWithAngles_44504451121313
// MI455X (gfx1250) — matrix operand order repaired and checked by reading, not compiled, not run on hardware
//
#include <hip/hip_runtime.h>
#include <math.h>

// ---------------- problem constants ----------------
#define NN      50000      // nodes
#define NE      800000     // edges
#define INC0    64
#define HIDC    128
#define OUTC    64
// padded edge-MLP widths (multiple of 32 = WMMA K)
#define DINP0   96         // 64+3=67 -> 96
#define DINP1   160        // 128+3=131 -> 160

typedef __attribute__((ext_vector_type(8)))  _Float16 v8h;
typedef __attribute__((ext_vector_type(16))) _Float16 v16h;
typedef __attribute__((ext_vector_type(8)))  float    v8f;
typedef __attribute__((__vector_size__(4 * sizeof(int)))) int v4i;

// ---- optional CDNA5 async global->LDS path ----
#if defined(__has_builtin)
#  if __has_builtin(__builtin_amdgcn_global_load_async_to_lds_b128) && \
      __has_builtin(__builtin_amdgcn_s_wait_asynccnt)
#    define HAVE_ASYNC 1
#  endif
#endif
#ifndef HAVE_ASYNC
#  define HAVE_ASYNC 0
#endif

__device__ __forceinline__ void copy16B_to_lds(const _Float16* g, _Float16* l) {
#if HAVE_ASYNC
  __builtin_amdgcn_global_load_async_to_lds_b128(
      (__attribute__((address_space(1))) v4i*)g,
      (__attribute__((address_space(3))) v4i*)l, 0, 0);
#else
  *(v8h*)l = *(const v8h*)g;
#endif
}
__device__ __forceinline__ void async_fence() {
#if HAVE_ASYNC
  __builtin_amdgcn_s_wait_asynccnt(0);
#endif
}

// ---- WMMA fragment loaders (wave32, contiguous k = 16*(lane/16) + s, same in A and B) ----
// A: 16x32 f16. lanes 0-15: row=lane,   K 0..15
//               lanes16-31: row=lane-16,K 16..31
__device__ __forceinline__ v16h load_a_frag(const _Float16* H, int stride,
                                            int m0, int k0, int lane) {
  int r  = lane & 15;
  int hs = lane >> 4;              // 0 or 1
  const _Float16* p = H + (size_t)(m0 + r) * stride + k0 + hs * 16;
  v8h lo = *(const v8h*)p;         // K k0+hs*16  .. +7
  v8h hi = *(const v8h*)(p + 8);   // K k0+hs*16+8.. +15
  return __builtin_shufflevector(lo, hi, 0,1,2,3,4,5,6,7,8,9,10,11,12,13,14,15);
}
// B: 32x16 f16 stored as W[n][k] row-major. lanes 0-15: col=lane, K 0..15;
// lanes 16-31: col=lane-16, K 16..31 (contiguous in k)
__device__ __forceinline__ v16h load_b_frag(const _Float16* W, int stride,
                                            int n0, int k0, int lane) {
  int n  = lane & 15;
  int ks = (lane >> 4) * 16;
  return *(const v16h*)(W + (size_t)(n0 + n) * stride + k0 + ks);
}

__device__ __forceinline__ v8f wmma_f16(v16h a, v16h b, v8f c) {
  return __builtin_amdgcn_wmma_f32_16x16x32_f16(false, a, false, b, (short)0, c,
                                                false, false);
}

// ---------------- prep kernels ----------------
// weight-norm + pad + fp32->fp16: Wp[n][k] = g[n]*v[n][k]/||v[n]|| (zero pad)
__global__ void wn_pad_kernel(const float* __restrict__ v, const float* __restrict__ g,
                              _Float16* __restrict__ Wp, int din, int dinP) {
  int n = blockIdx.x;
  __shared__ float red[64];
  float s = 0.f;
  if (n < din)
    for (int k = threadIdx.x; k < din; k += 64) { float t = v[(size_t)n*din + k]; s += t*t; }
  red[threadIdx.x] = s;
  __syncthreads();
  for (int off = 32; off > 0; off >>= 1) {
    if (threadIdx.x < off) red[threadIdx.x] += red[threadIdx.x + off];
    __syncthreads();
  }
  float scale = (n < din) ? g[n] * rsqrtf(fmaxf(red[0], 1e-30f)) : 0.f;
  for (int k = threadIdx.x; k < dinP; k += 64)
    Wp[(size_t)n*dinP + k] = (_Float16)((n < din && k < din) ? v[(size_t)n*din + k] * scale : 0.f);
}

__global__ void pad_f16_kernel(const float* __restrict__ W, _Float16* __restrict__ Wp,
                               int rows, int cols, int rowsP, int colsP) {
  int idx = blockIdx.x * blockDim.x + threadIdx.x;
  if (idx >= rowsP * colsP) return;
  int r = idx / colsP, c = idx % colsP;
  Wp[idx] = (_Float16)((r < rows && c < cols) ? W[(size_t)r*cols + c] : 0.f);
}

__global__ void pad_f32_kernel(const float* __restrict__ b, float* __restrict__ bp,
                               int n, int nP) {
  int i = blockIdx.x * blockDim.x + threadIdx.x;
  if (i < nP) bp[i] = (i < n) ? b[i] : 0.f;
}

__global__ void zero_kernel(float* __restrict__ p, int n) {
  int i = blockIdx.x * blockDim.x + threadIdx.x;
  if (i < n) p[i] = 0.f;
}

// ---------------- edge MLP (WMMA) ----------------
// 128 edges / block, 8 waves, each wave owns 16 edge rows.
template<int INC, int DINP, int OUTP, bool LASTGNN>
__global__ __launch_bounds__(256)
void edge_mlp_kernel(const float* __restrict__ xin,   // [NN, INC] f32
                     const float* __restrict__ ang,   // [NE] degrees
                     const float* __restrict__ dist,  // [NE]
                     const int*   __restrict__ srcI,  // [NE]
                     const int*   __restrict__ dstI,  // [NE]
                     const _Float16* __restrict__ Whid,   // 4*DINP*DINP
                     const _Float16* __restrict__ Wout,   // OUTP*DINP
                     const float* __restrict__ biasHid,   // 4*DINP (padded)
                     const float* __restrict__ biasOut,   // OUTP
                     float* __restrict__ sum,             // [NN, OUTP]
                     float* __restrict__ cnt,             // [NN]
                     int nEdges) {
  constexpr int WROWS = (OUTP > DINP) ? OUTP : DINP;
  constexpr int KT    = DINP / 32;
  __shared__ _Float16 Hbuf[2][128 * DINP];
  __shared__ _Float16 Wlds[WROWS * DINP];
  __shared__ float    biasLds[4 * DINP + OUTP];
  __shared__ int srcLds[128];
  __shared__ int dstLds[128];

  const int tid  = threadIdx.x;
  const int lane = tid & 31;
  const int wave = tid >> 5;
  const int m0   = wave * 16;
  const int tile = blockIdx.x * 128;

  if (tid < 128) {
    int e = tile + tid;
    srcLds[tid] = (e < nEdges) ? srcI[e] : 0;
    dstLds[tid] = (e < nEdges) ? dstI[e] : 0;
  }
  for (int i = tid; i < 4 * DINP; i += 256) biasLds[i] = biasHid[i];
  for (int i = tid; i < OUTP; i += 256)     biasLds[4 * DINP + i] = biasOut[i];
  __syncthreads();

  // stage edge features -> Hbuf[0] (f16); zero rows for OOB edges / pad cols
  const float d2r = 0.017453292519943295f;
  for (int idx = tid; idx < 128 * DINP; idx += 256) {
    int r = idx / DINP, c = idx % DINP;
    int e = tile + r;
    float v = 0.f;
    if (e < nEdges) {
      if (c < INC)            v = xin[(size_t)srcLds[r] * INC + c];
      else if (c == INC)      v = sinf(ang[e] * d2r);
      else if (c == INC + 1)  v = cosf(ang[e] * d2r);
      else if (c == INC + 2)  v = dist[e];
    }
    Hbuf[0][idx] = (_Float16)v;
  }

  int cur = 0;
  // 4 weight-normed square linears (pure linear chain in the reference)
  for (int s = 0; s < 4; ++s) {
    const _Float16* Wg = Whid + (size_t)s * DINP * DINP;
    if (s < 3) __builtin_prefetch(Whid + (size_t)(s + 1) * DINP * DINP, 0, 1);
    for (int idx = tid * 8; idx < DINP * DINP; idx += 256 * 8)
      copy16B_to_lds(&Wg[idx], &Wlds[idx]);
    async_fence();
    __syncthreads();

    const _Float16* Hin  = &Hbuf[cur][0];
    _Float16*       Hout = &Hbuf[cur ^ 1][0];

    // hoist this wave's 16 activation rows into registers (reused for all N-tiles)
    v16h afr[KT];
#pragma unroll
    for (int kt = 0; kt < KT; ++kt)
      afr[kt] = load_a_frag(Hin, DINP, m0, kt * 32, lane);

    for (int nt = 0; nt < DINP / 16; nt += 2) {   // two independent acc tiles
      v8f acc0 = {}, acc1 = {};
#pragma unroll
      for (int kt = 0; kt < KT; ++kt) {
        v16h b0 = load_b_frag(Wlds, DINP, nt * 16,       kt * 32, lane);
        v16h b1 = load_b_frag(Wlds, DINP, (nt + 1) * 16, kt * 32, lane);
        acc0 = wmma_f16(afr[kt], b0, acc0);
        acc1 = wmma_f16(afr[kt], b1, acc1);
      }
      // D layout: col = lane&15, rows m0+(lane>>4)*8 .. +7
      int nc0  = nt * 16 + (lane & 15);
      int nc1  = nc0 + 16;
      int mrow = m0 + (lane >> 4) * 8;
      float b0 = biasLds[s * DINP + nc0];
      float b1 = biasLds[s * DINP + nc1];
#pragma unroll
      for (int i = 0; i < 8; ++i) {
        Hout[(size_t)(mrow + i) * DINP + nc0] = (_Float16)(acc0[i] + b0);
        Hout[(size_t)(mrow + i) * DINP + nc1] = (_Float16)(acc1[i] + b1);
      }
    }
    __syncthreads();
    cur ^= 1;
  }

  // final projection: [*,DINP] x [DINP,OUTP] + bias (+ exact GELU if not last GNN layer)
  for (int idx = tid * 8; idx < OUTP * DINP; idx += 256 * 8)
    copy16B_to_lds(&Wout[idx], &Wlds[idx]);
  async_fence();
  __syncthreads();

  const _Float16* Hin = &Hbuf[cur][0];
  v16h afr[KT];
#pragma unroll
  for (int kt = 0; kt < KT; ++kt)
    afr[kt] = load_a_frag(Hin, DINP, m0, kt * 32, lane);

  for (int nt = 0; nt < OUTP / 16; nt += 2) {
    v8f acc0 = {}, acc1 = {};
#pragma unroll
    for (int kt = 0; kt < KT; ++kt) {
      v16h b0 = load_b_frag(Wlds, DINP, nt * 16,       kt * 32, lane);
      v16h b1 = load_b_frag(Wlds, DINP, (nt + 1) * 16, kt * 32, lane);
      acc0 = wmma_f16(afr[kt], b0, acc0);
      acc1 = wmma_f16(afr[kt], b1, acc1);
    }
    int nc0  = nt * 16 + (lane & 15);
    int nc1  = nc0 + 16;
    int mrow = m0 + (lane >> 4) * 8;
    float b0 = biasLds[4 * DINP + nc0];
    float b1 = biasLds[4 * DINP + nc1];
#pragma unroll
    for (int i = 0; i < 8; ++i) {
      int e = tile + mrow + i;
      if (e < nEdges) {
        float v0 = acc0[i] + b0;
        float v1 = acc1[i] + b1;
        if (!LASTGNN) {
          v0 = 0.5f * v0 * (1.f + erff(v0 * 0.70710678118654752f));  // exact GELU
          v1 = 0.5f * v1 * (1.f + erff(v1 * 0.70710678118654752f));
        }
        float* row = &sum[(size_t)dstLds[mrow + i] * OUTP];
        atomicAdd(row + nc0, v0);
        atomicAdd(row + nc1, v1);
      }
    }
  }
  if (tid < 128) {
    int e = tile + tid;
    if (e < nEdges) atomicAdd(&cnt[dstLds[tid]], 1.0f);
  }
}

// ---------------- node update: mean -> W_agg GEMV -> biases (-> tanh) ----------------
template<int OC, bool APPLY_TANH>
__global__ __launch_bounds__(128)
void node_update_kernel(const float* __restrict__ sum, const float* __restrict__ cnt,
                        const float* __restrict__ Wagg, const float* __restrict__ bagg,
                        const float* __restrict__ bias, float* __restrict__ out, int nNodes) {
  __shared__ float Wl[OC * OC];
  __shared__ float meanL[OC];
  for (int i = threadIdx.x; i < OC * OC; i += blockDim.x) Wl[i] = Wagg[i];
  for (int n = blockIdx.x; n < nNodes; n += gridDim.x) {
    __syncthreads();
    float c = fmaxf(cnt[n], 1.0f);
    for (int k = threadIdx.x; k < OC; k += blockDim.x)
      meanL[k] = sum[(size_t)n * OC + k] / c;
    __syncthreads();
    for (int o = threadIdx.x; o < OC; o += blockDim.x) {
      float acc = bagg[o] + bias[o];
#pragma unroll 8
      for (int k = 0; k < OC; ++k) acc += meanL[k] * Wl[o * OC + k];
      out[(size_t)n * OC + o] = APPLY_TANH ? tanhf(acc) : acc;
    }
  }
}

// ---------------- launch ----------------
extern "C" void kernel_launch(void* const* d_in, const int* in_sizes, int n_in,
                              void* d_out, int out_size, void* d_ws, size_t ws_size,
                              hipStream_t stream) {
  // input order: x, edge_attr, edge_distance,
  //  layer0: v0..v3, g0..g3, b0..b3, W_out, b_out, W_agg, b_agg, bias  (idx 3..19)
  //  layer1: same                                                      (idx 20..36)
  //  edge_index (2 x NE)                                               (idx 37)
  const float* x    = (const float*)d_in[0];
  const float* ang  = (const float*)d_in[1];
  const float* dist = (const float*)d_in[2];
  const int*   ei   = (const int*)d_in[37];
  const int*   src  = ei;
  const int*   dst  = ei + NE;

  // workspace carve-up
  auto al = [](size_t x) { return (x + 255) & ~(size_t)255; };
  char* ws = (char*)d_ws;
  size_t off = 0;
  float*    h0       = (float*)(ws + off); off = al(off + (size_t)NN * HIDC * 4);
  float*    sum      = (float*)(ws + off); off = al(off + (size_t)NN * HIDC * 4);
  float*    cnt      = (float*)(ws + off); off = al(off + (size_t)NN * 4);
  _Float16* WhL0     = (_Float16*)(ws + off); off = al(off + (size_t)4 * DINP0 * DINP0 * 2);
  _Float16* WoutL0   = (_Float16*)(ws + off); off = al(off + (size_t)HIDC * DINP0 * 2);
  float*    bHidL0   = (float*)(ws + off); off = al(off + (size_t)4 * DINP0 * 4);
  float*    bOutL0   = (float*)(ws + off); off = al(off + (size_t)HIDC * 4);
  _Float16* WhL1     = (_Float16*)(ws + off); off = al(off + (size_t)4 * DINP1 * DINP1 * 2);
  _Float16* WoutL1   = (_Float16*)(ws + off); off = al(off + (size_t)OUTC * DINP1 * 2);
  float*    bHidL1   = (float*)(ws + off); off = al(off + (size_t)4 * DINP1 * 4);
  float*    bOutL1   = (float*)(ws + off); off = al(off + (size_t)OUTC * 4);
  (void)ws_size; (void)in_sizes; (void)n_in; (void)out_size;

  const int din0 = INC0 + 3;   // 67
  const int din1 = HIDC + 3;   // 131

  // ---- prep layer 0 ----
  for (int s = 0; s < 4; ++s) {
    wn_pad_kernel<<<DINP0, 64, 0, stream>>>((const float*)d_in[3 + s], (const float*)d_in[7 + s],
                                            WhL0 + (size_t)s * DINP0 * DINP0, din0, DINP0);
    pad_f32_kernel<<<(DINP0 + 255) / 256, 256, 0, stream>>>((const float*)d_in[11 + s],
                                                            bHidL0 + s * DINP0, din0, DINP0);
  }
  pad_f16_kernel<<<(HIDC * DINP0 + 255) / 256, 256, 0, stream>>>((const float*)d_in[15], WoutL0,
                                                                 HIDC, din0, HIDC, DINP0);
  pad_f32_kernel<<<(HIDC + 255) / 256, 256, 0, stream>>>((const float*)d_in[16], bOutL0, HIDC, HIDC);

  // ---- prep layer 1 ----
  for (int s = 0; s < 4; ++s) {
    wn_pad_kernel<<<DINP1, 64, 0, stream>>>((const float*)d_in[20 + s], (const float*)d_in[24 + s],
                                            WhL1 + (size_t)s * DINP1 * DINP1, din1, DINP1);
    pad_f32_kernel<<<(DINP1 + 255) / 256, 256, 0, stream>>>((const float*)d_in[28 + s],
                                                            bHidL1 + s * DINP1, din1, DINP1);
  }
  pad_f16_kernel<<<(OUTC * DINP1 + 255) / 256, 256, 0, stream>>>((const float*)d_in[32], WoutL1,
                                                                 OUTC, din1, OUTC, DINP1);
  pad_f32_kernel<<<(OUTC + 255) / 256, 256, 0, stream>>>((const float*)d_in[33], bOutL1, OUTC, OUTC);

  const int nEdgeBlocks = (NE + 127) / 128;

  // ---- GNN layer 0 ----
  zero_kernel<<<((NN * HIDC) + 255) / 256, 256, 0, stream>>>(sum, NN * HIDC);
  zero_kernel<<<(NN + 255) / 256, 256, 0, stream>>>(cnt, NN);
  edge_mlp_kernel<INC0, DINP0, HIDC, false><<<nEdgeBlocks, 256, 0, stream>>>(
      x, ang, dist, src, dst, WhL0, WoutL0, bHidL0, bOutL0, sum, cnt, NE);
  node_update_kernel<HIDC, true><<<1024, 128, 0, stream>>>(
      sum, cnt, (const float*)d_in[17], (const float*)d_in[18], (const float*)d_in[19], h0, NN);

  // ---- GNN layer 1 (final) ----
  zero_kernel<<<((NN * OUTC) + 255) / 256, 256, 0, stream>>>(sum, NN * OUTC);
  zero_kernel<<<(NN + 255) / 256, 256, 0, stream>>>(cnt, NN);
  edge_mlp_kernel<HIDC, DINP1, OUTC, true><<<nEdgeBlocks, 256, 0, stream>>>(
      h0, ang, dist, src, dst, WhL1, WoutL1, bHidL1, bOutL1, sum, cnt, NE);
  node_update_kernel<OUTC, false><<<1024, 128, 0, stream>>>(
      sum, cnt, (const float*)d_in[34], (const float*)d_in[35], (const float*)d_in[36],
      (float*)d_out, NN);
}
